// CapsDiscriminatorMNIST_36043365548420
// MI455X (gfx1250) — matrix operand order repaired and checked by reading, not compiled, not run on hardware
//
#include <hip/hip_runtime.h>
#include <stdint.h>
#include <math.h>

typedef _Float16 h16;
typedef __attribute__((ext_vector_type(16))) _Float16 v16h;
typedef __attribute__((ext_vector_type(8)))  float    v8f;

#define TPB 256

// ---------------- problem sizes ----------------
#define BATCH 512
#define C1    256            // conv1 out channels
#define P1    400            // 20*20 spatial per image
#define K1    81
#define K1P   96             // padded K for conv1 GEMM
#define M1    (BATCH*P1)     // 204800

#define C2    256
#define P2    36             // 6*6
#define K2    (256*81)       // 20736, iterated as k = kq*256 + c (kq = ky*9+kx)
#define M2    (BATCH*P2)     // 18432

// ---------------- GEMM tiling ----------------
#define MT  128              // block tile rows
#define NT  64               // block tile cols
#define KT  32
#define LDA 40               // halves per LDS row (pad 32 -> 40 to spread banks)

// ---- CDNA5 async global->LDS (ASYNCcnt) helpers ----
__device__ __forceinline__ void async_b128(uint32_t ldsOff, uint64_t gaddr) {
  asm volatile("global_load_async_to_lds_b128 %0, %1, off" :: "v"(ldsOff), "v"(gaddr) : "memory");
}
#define WAIT_ASYNC_3() asm volatile("s_wait_asynccnt 0x3" ::: "memory")
#define WAIT_ASYNC_0() asm volatile("s_wait_asynccnt 0x0" ::: "memory")

// =============== helper / conversion kernels ===============
__global__ void build_w1t_kernel(const float* __restrict__ w, h16* __restrict__ dst) {
  int i = blockIdx.x * TPB + threadIdx.x;          // over 256*96
  if (i >= C1 * K1P) return;
  int n = i / K1P, k = i - n * K1P;
  dst[i] = (k < K1) ? (h16)w[n * K1 + k] : (h16)0.f;
}

// permute pc_w [n][c][ky][kx] -> w2t [n][kq=ky*9+kx][c] (f16): 32 consecutive K
// elements = 32 consecutive channels (contiguous in NHWC h1)
__global__ void build_w2t_kernel(const float* __restrict__ pc_w, h16* __restrict__ dst) {
  int i = blockIdx.x * TPB + threadIdx.x;          // over 256*20736
  if (i >= C2 * K2) return;
  int n = i / K2; int kp = i - n * K2;
  int kq = kp >> 8; int c = kp & 255;
  dst[i] = (h16)pc_w[((size_t)n * 256 + c) * K1 + kq];
}

// materialize conv1 im2col: A1[m][96] f16 (zero-padded k>=81); removes all
// branches/divisions from the conv1 GEMM hot loop
__global__ void im2col_x1_kernel(const float* __restrict__ x, h16* __restrict__ A1) {
  int i = blockIdx.x * TPB + threadIdx.x;          // over M1*K1P = 19,660,800
  if (i >= M1 * K1P) return;
  int m = i / K1P; int k = i - m * K1P;
  int bi = m / P1; int p = m - bi * P1;
  int oy = p / 20; int ox = p - oy * 20;
  h16 v = (h16)0.f;
  if (k < K1) { int ky = k / 9, kx = k - ky * 9;
                v = (h16)x[bi * 784 + (oy + ky) * 28 + (ox + kx)]; }
  A1[i] = v;
}

// fragment builders: CDNA5 wave32 WMMA VGPR layouts
// A 16x32 f16: lane m=lane%16, grp=lane/16; vgpr v holds k = 2(v%4)+8*grp+16*(v/4)
// B 32x16 f16: lane n=lane%16;             vgpr v holds k = 2(v%4)+8*grp+16*(v/4)
__device__ __forceinline__ v16h frag_a(const h16* As, int rowBase, int lm, int grp) {
  v16h a;
#pragma unroll
  for (int v = 0; v < 8; ++v) {
    int k = ((v & 3) << 1) + (grp << 3) + ((v >> 2) << 4);
    union { uint32_t u; h16 h[2]; } c;
    c.u = *(const uint32_t*)&As[(rowBase + lm) * LDA + k];
    a[2 * v] = c.h[0]; a[2 * v + 1] = c.h[1];
  }
  return a;
}
__device__ __forceinline__ v16h frag_b(const h16* Bs, int colBase, int lm, int grp) {
  v16h b;
#pragma unroll
  for (int v = 0; v < 8; ++v) {
    int k = ((v & 3) << 1) + (grp << 3) + ((v >> 2) << 4);
    union { uint32_t u; h16 h[2]; } c;
    c.u = *(const uint32_t*)&Bs[(colBase + lm) * LDA + k];
    b[2 * v] = c.h[0]; b[2 * v + 1] = c.h[1];
  }
  return b;
}
#define WMMA(acc, a, b) \
  acc = __builtin_amdgcn_wmma_f32_16x16x32_f16(false, a, false, b, (short)0, acc, false, false)

// =============== conv1 as GEMM (WMMA f16, fully async-staged) ===============
// A1 [M1 x 96] x w1t [256 x 96]^T -> h1 NHWC f16 = relu(acc + bias)
__global__ __launch_bounds__(TPB) void conv1_wmma(const h16* __restrict__ A1,
                                                  const h16* __restrict__ w1t,
                                                  const float* __restrict__ bias,
                                                  h16* __restrict__ h1) {
  __shared__ __align__(16) h16 As[2][MT * LDA];
  __shared__ __align__(16) h16 Bs[2][NT * LDA];
  const int tid   = threadIdx.x;
  const int tileM = blockIdx.x % (M1 / MT);
  const int tileN = blockIdx.x / (M1 / MT);
  const int mBase = tileM * MT, nBase = tileN * NT;

  const int aRow = tid & 127, aSeg = tid >> 7;     // A: 128 rows x 2 segs of 16 k
  const int bRow = tid >> 2,  bSeg = tid & 3;      // B: 64 rows x 4 segs of 8 k
  const int lane = tid & 31, lm = lane & 15, grp = lane >> 4;
  const int wv = tid >> 5;                         // 8 waves: 4 (m) x 2 (n)
  const int wm = wv & 3, wn = wv >> 2;

  const uint64_t gaA0 = (uint64_t)(uintptr_t)(A1 + (size_t)(mBase + aRow) * K1P + aSeg * 16);
  const uint64_t gaB0 = (uint64_t)(uintptr_t)(w1t + (size_t)(nBase + bRow) * K1P + bSeg * 8);
  auto issue = [&](int kb, int buf) {              // 3 async b128 per thread per step
    uint32_t la = (uint32_t)(uintptr_t)&As[buf][aRow * LDA + aSeg * 16];
    async_b128(la,      gaA0 + (size_t)kb * 2);
    async_b128(la + 16, gaA0 + (size_t)kb * 2 + 16);
    uint32_t lb = (uint32_t)(uintptr_t)&Bs[buf][bRow * LDA + bSeg * 8];
    async_b128(lb, gaB0 + (size_t)kb * 2);
  };

  v8f acc00 = {}, acc01 = {}, acc10 = {}, acc11 = {};
  issue(0, 0);
  for (int kb = 0; kb < K1P; kb += KT) {
    const int buf = (kb >> 5) & 1;
    if (kb + KT < K1P) { issue(kb + KT, buf ^ 1); WAIT_ASYNC_3(); }
    else              { WAIT_ASYNC_0(); }
    __syncthreads();                               // all waves' tiles for `buf` landed
    v16h a0 = frag_a(As[buf], wm * 32,      lm, grp);
    v16h a1 = frag_a(As[buf], wm * 32 + 16, lm, grp);
    v16h b0 = frag_b(Bs[buf], wn * 32,      lm, grp);
    v16h b1 = frag_b(Bs[buf], wn * 32 + 16, lm, grp);
    WMMA(acc00, a0, b0); WMMA(acc01, a0, b1);
    WMMA(acc10, a1, b0); WMMA(acc11, a1, b1);
    __syncthreads();                               // done reading `buf` before it is re-filled
  }
  const int n0 = nBase + wn * 32 + lm;
  const float bv0 = bias[n0], bv1 = bias[n0 + 16];
#pragma unroll
  for (int v = 0; v < 8; ++v) {
    int mm0 = mBase + wm * 32 + (grp << 3) + v;
    int mm1 = mm0 + 16;
    int b0i = mm0 / P1, p0 = mm0 - b0i * P1;
    int b1i = mm1 / P1, p1 = mm1 - b1i * P1;
    h1[((size_t)b0i * P1 + p0) * C1 + n0]      = (h16)fmaxf(acc00[v] + bv0, 0.f);
    h1[((size_t)b0i * P1 + p0) * C1 + n0 + 16] = (h16)fmaxf(acc01[v] + bv1, 0.f);
    h1[((size_t)b1i * P1 + p1) * C1 + n0]      = (h16)fmaxf(acc10[v] + bv0, 0.f);
    h1[((size_t)b1i * P1 + p1) * C1 + n0 + 16] = (h16)fmaxf(acc11[v] + bv1, 0.f);
  }
}

// =============== primary-caps conv as implicit GEMM (WMMA f16, async) ===============
// K ordered (kq, c): A[m][k] = h1_nhwc[b][(2oy+ky)*20 + 2ox+kx][c] (contiguous in c)
// B: w2t [n][kq][c].  C: s2[b][n*36+p] f32 (+bias) (== the (B,1152,8) reshape order)
__global__ __launch_bounds__(TPB) void conv2_wmma(const h16* __restrict__ h1,
                                                  const h16* __restrict__ w2t,
                                                  const float* __restrict__ bias,
                                                  float* __restrict__ s2) {
  __shared__ __align__(16) h16 As[2][MT * LDA];
  __shared__ __align__(16) h16 Bs[2][NT * LDA];
  const int tid   = threadIdx.x;
  const int tileM = blockIdx.x % (M2 / MT);
  const int tileN = blockIdx.x / (M2 / MT);
  const int mBase = tileM * MT, nBase = tileN * NT;

  const int aRow = tid & 127, aSeg = tid >> 7;
  const int m  = mBase + aRow;
  const int bi = m / P2; const int p = m - bi * P2;
  const int oy = p / 6;  const int ox = p - oy * 6;
  const h16* hrow = h1 + ((size_t)bi * P1 + (2 * oy) * 20 + 2 * ox) * C1;  // + (ky*20+kx)*256 + c
  const int bRow = tid >> 2, bSeg = tid & 3;

  const int lane = tid & 31, lm = lane & 15, grp = lane >> 4;
  const int wv = tid >> 5;
  const int wm = wv & 3, wn = wv >> 2;

  const uint64_t gaB0 = (uint64_t)(uintptr_t)(w2t + (size_t)(nBase + bRow) * K2 + bSeg * 8);
  auto issue = [&](int kb, int buf) {              // kb = kq*256 + c0
    int kq = kb >> 8, c0 = kb & 255;
    int ky = kq / 9, kx = kq - ky * 9;
    uint64_t gaA = (uint64_t)(uintptr_t)(hrow + (ky * 20 + kx) * C1 + c0 + aSeg * 16);
    uint32_t la = (uint32_t)(uintptr_t)&As[buf][aRow * LDA + aSeg * 16];
    async_b128(la,      gaA);
    async_b128(la + 16, gaA + 16);
    uint32_t lb = (uint32_t)(uintptr_t)&Bs[buf][bRow * LDA + bSeg * 8];
    async_b128(lb, gaB0 + (size_t)kb * 2);
  };

  v8f acc00 = {}, acc01 = {}, acc10 = {}, acc11 = {};
  issue(0, 0);
  for (int kb = 0; kb < K2; kb += KT) {
    const int buf = (kb >> 5) & 1;
    if (kb + KT < K2) { issue(kb + KT, buf ^ 1); WAIT_ASYNC_3(); }
    else             { WAIT_ASYNC_0(); }
    __syncthreads();
    v16h a0 = frag_a(As[buf], wm * 32,      lm, grp);
    v16h a1 = frag_a(As[buf], wm * 32 + 16, lm, grp);
    v16h b0 = frag_b(Bs[buf], wn * 32,      lm, grp);
    v16h b1 = frag_b(Bs[buf], wn * 32 + 16, lm, grp);
    WMMA(acc00, a0, b0); WMMA(acc01, a0, b1);
    WMMA(acc10, a1, b0); WMMA(acc11, a1, b1);
    __syncthreads();
  }
  const int n0 = nBase + wn * 32 + lm;
  const float bv0 = bias[n0], bv1 = bias[n0 + 16];
#pragma unroll
  for (int v = 0; v < 8; ++v) {
    int mm0 = mBase + wm * 32 + (grp << 3) + v;
    int mm1 = mm0 + 16;
    int b0i = mm0 / P2, p0 = mm0 - b0i * P2;
    int b1i = mm1 / P2, p1 = mm1 - b1i * P2;
    s2[(size_t)b0i * 9216 + n0 * P2 + p0]        = acc00[v] + bv0;
    s2[(size_t)b0i * 9216 + (n0 + 16) * P2 + p0] = acc01[v] + bv1;
    s2[(size_t)b1i * 9216 + n0 * P2 + p1]        = acc10[v] + bv0;
    s2[(size_t)b1i * 9216 + (n0 + 16) * P2 + p1] = acc11[v] + bv1;
  }
}

// =============== squash primary capsules ===============
__global__ void squash_caps(const float* __restrict__ s2, float* __restrict__ u) {
  int idx = blockIdx.x * TPB + threadIdx.x;          // capsule id over 512*1152
  if (idx >= BATCH * 1152) return;
  const float* p = s2 + (size_t)idx * 8;
  float* q = u + (size_t)idx * 8;
  float v[8]; float ss = 0.f;
#pragma unroll
  for (int d = 0; d < 8; ++d) { v[d] = p[d]; ss += v[d] * v[d]; }
  float nrm = sqrtf(ss);
  float sc = ss / (1.f + ss) / (nrm + 1e-8f);
#pragma unroll
  for (int d = 0; d < 8; ++d) q[d] = sc * v[d];
}

// =============== fused dynamic routing (one block per batch element) ===============
// Recomputes votes x_hat on the fly from caps_w (L2-resident, 5.9MB) instead of
// materializing the 377MB x_hat tensor. u, b_logits, c live in LDS (~130KB of 320KB WGP LDS).
__global__ __launch_bounds__(TPB) void routing_kernel(const float* __restrict__ u,
                                                      const float* __restrict__ caps_w,
                                                      float* __restrict__ out) {
  __shared__ float u_l[1152 * 8];
  __shared__ float blog[10 * 1152];
  __shared__ float c_l[10 * 1152];
  __shared__ float s_l[160];
  __shared__ float v_l[160];
  __shared__ float scale_l[10];
  const int tid = threadIdx.x;
  const int b = blockIdx.x;

  for (int idx = tid; idx < 9216; idx += TPB)  u_l[idx] = u[(size_t)b * 9216 + idx];
  for (int idx = tid; idx < 11520; idx += TPB) blog[idx] = 0.f;
  __syncthreads();

  for (int it = 0; it < 3; ++it) {
    for (int i = tid; i < 1152; i += TPB) {
      float mx = -1e30f;
#pragma unroll
      for (int o = 0; o < 10; ++o) mx = fmaxf(mx, blog[o * 1152 + i]);
      float sum = 0.f;
#pragma unroll
      for (int o = 0; o < 10; ++o) { float e = __expf(blog[o * 1152 + i] - mx); c_l[o * 1152 + i] = e; sum += e; }
      float inv = 1.f / sum;
#pragma unroll
      for (int o = 0; o < 10; ++o) c_l[o * 1152 + i] *= inv;
    }
    __syncthreads();
    if (tid < 160) {                       // s[o][d] = sum_i c[o][i]*(caps_w[o,i,d,:].u[i,:])
      int o = tid >> 4, d = tid & 15;
      float s = 0.f;
      for (int i = 0; i < 1152; ++i) {
        const float* w  = caps_w + (((size_t)(o * 1152 + i) * 16 + d) << 3);
        const float* ui = &u_l[i * 8];
        float xh = w[0]*ui[0] + w[1]*ui[1] + w[2]*ui[2] + w[3]*ui[3]
                 + w[4]*ui[4] + w[5]*ui[5] + w[6]*ui[6] + w[7]*ui[7];
        s += c_l[o * 1152 + i] * xh;
      }
      s_l[tid] = s;
    }
    __syncthreads();
    if (tid < 10) {
      float ss = 0.f;
#pragma unroll
      for (int d = 0; d < 16; ++d) ss += s_l[tid * 16 + d] * s_l[tid * 16 + d];
      float nrm = sqrtf(ss);
      scale_l[tid] = ss / (1.f + ss) / (nrm + 1e-8f);
    }
    __syncthreads();
    if (tid < 160) v_l[tid] = scale_l[tid >> 4] * s_l[tid];
    __syncthreads();
    if (it == 2) break;
    for (int i = tid; i < 1152; i += TPB) {  // blog[o][i] += v[o][:] . xhat[o][i][:]
      const float* ui = &u_l[i * 8];
      for (int o = 0; o < 10; ++o) {
        const float* w = caps_w + ((size_t)(o * 1152 + i) << 7);
        float acc = 0.f;
#pragma unroll
        for (int d = 0; d < 16; ++d) {
          const float* wd = w + d * 8;
          float xh = wd[0]*ui[0] + wd[1]*ui[1] + wd[2]*ui[2] + wd[3]*ui[3]
                   + wd[4]*ui[4] + wd[5]*ui[5] + wd[6]*ui[6] + wd[7]*ui[7];
          acc += v_l[o * 16 + d] * xh;
        }
        blog[o * 1152 + i] += acc;
      }
    }
    __syncthreads();
  }
  if (tid < 10) {
    float ss = 0.f;
#pragma unroll
    for (int d = 0; d < 16; ++d) ss += v_l[tid * 16 + d] * v_l[tid * 16 + d];
    out[(size_t)b * 10 + tid] = sqrtf(ss);
  }
}

// =============== launch ===============
extern "C" void kernel_launch(void* const* d_in, const int* in_sizes, int n_in,
                              void* d_out, int out_size, void* d_ws, size_t ws_size,
                              hipStream_t stream) {
  const float* x       = (const float*)d_in[0];
  const float* conv1_w = (const float*)d_in[1];
  const float* conv1_b = (const float*)d_in[2];
  const float* pc_w    = (const float*)d_in[3];
  const float* pc_b    = (const float*)d_in[4];
  const float* caps_w  = (const float*)d_in[5];
  float* out = (float*)d_out;

  char* ws = (char*)d_ws;
  size_t off = 0;
  auto take = [&](size_t bytes) -> char* {
    char* p = ws + off; off += (bytes + 255) & ~(size_t)255; return p;
  };
  h16*   A1  = (h16*)take((size_t)M1 * K1P * 2);              // 39.3 MB im2col(x)
  h16*   w1t = (h16*)take((size_t)C1 * K1P * 2);              // 48 KB
  h16*   w2t = (h16*)take((size_t)C2 * K2 * 2);               // 10.6 MB
  h16*   h1  = (h16*)take((size_t)BATCH * P1 * C1 * 2);       // 105 MB NHWC (L2-resident)
  float* s2  = (float*)take((size_t)BATCH * 9216 * 4);        // 18.9 MB
  float* u   = (float*)take((size_t)BATCH * 9216 * 4);        // 18.9 MB

  im2col_x1_kernel<<<(M1 * K1P + TPB - 1) / TPB, TPB, 0, stream>>>(x, A1);
  build_w1t_kernel<<<(C1 * K1P + TPB - 1) / TPB, TPB, 0, stream>>>(conv1_w, w1t);
  build_w2t_kernel<<<(C2 * K2 + TPB - 1) / TPB, TPB, 0, stream>>>(pc_w, w2t);

  conv1_wmma<<<(M1 / MT) * (C1 / NT), TPB, 0, stream>>>(A1, w1t, conv1_b, h1);
  conv2_wmma<<<(M2 / MT) * (C2 / NT), TPB, 0, stream>>>(h1, w2t, pc_b, s2);

  squash_caps<<<(BATCH * 1152 + TPB - 1) / TPB, TPB, 0, stream>>>(s2, u);
  routing_kernel<<<BATCH, TPB, 0, stream>>>(u, caps_w, out);
}
